// LstmDecoder_45037027065874
// MI455X (gfx1250) — compile-verified, matrix operand order checked by reading, not run on hardware
//
#include <hip/hip_runtime.h>
#include <hip/hip_bf16.h>

// ---------------------------------------------------------------------------
// LSTM decoder for MI455X (gfx1250, wave32, WMMA).
// B=2048, H=256, 4H=1024, O=128, T=100.
//
// 128 workgroups x 256 threads (8 wave32s). Each WG owns 16 batch rows and
// runs the entire T=100 recurrence locally (batch rows are independent in the
// recurrence) -> no grid sync. Per step: [16,256]x[256,1024] bf16 WMMA GEMM,
// native-transcendental LSTM cell, [16,256]x[256,128] WMMA projection.
// W_hh fragments stream from L2 (512KB/step/WG, L2-resident); W_out and
// x_gates fragments are register-resident; h round-trips LDS as bf16.
// ---------------------------------------------------------------------------

typedef __bf16 v16bf __attribute__((ext_vector_type(16)));
typedef float  v8f   __attribute__((ext_vector_type(8)));
typedef int    i4    __attribute__((ext_vector_type(4)));

#define HH     256     // hidden
#define G4     1024    // 4*H
#define BATCH  2048
#define OUT    128
#define TSTEPS 100
#define PITCHH 264     // bf16 elems per LDS row: 528 bytes (16B aligned, bank-spread)

// D = A(16x32 bf16) x B(32x16 bf16) + C(16x16 f32)
static __device__ inline v8f wmma_bf16(v16bf a, v16bf b, v8f c) {
  return __builtin_amdgcn_wmma_f32_16x16x32_bf16(
      /*neg_a=*/false, a, /*neg_b=*/false, b,
      /*c_mod=*/(short)0, c, /*reuse_a=*/false, /*reuse_b=*/false);
}

// B-fragment (32x16, bf16): lane L holds column n = L%16; chunk q=0..7 covers
// k_local = 8*(L>>4) + 16*p + q. Packed per fragment as two coalesced 512B
// halves: part p at p*512 bytes, lane L's 16B chunk at lane*16.
static __device__ inline v16bf load_bfrag(const __bf16* frags, int fi, int lane) {
  const __bf16* base = frags + (size_t)fi * 512 + lane * 8;
  union { v16bf v; i4 q[2]; } u;
  u.q[0] = *(const i4*)(base);        // part 0
  u.q[1] = *(const i4*)(base + 256);  // part 1
  return u.v;
}

// A-fragment (16x32, bf16) from LDS row-major [16][PITCHH] bf16:
// lane L holds row m = L%16; VGPRs 0-3: k = kt*32 + 8*(L>>4) + 0..7,
// VGPRs 4-7: k = kt*32 + 16 + 8*(L>>4) + 0..7  -> two 16B ds loads.
static __device__ inline v16bf load_afrag(const __bf16* hlds, int ktile, int lane) {
  int m = lane & 15, hf = lane >> 4;
  const __bf16* base = hlds + m * PITCHH + ktile * 32 + 8 * hf;
  union { v16bf v; i4 q[2]; } u;
  u.q[0] = *(const i4*)(base);
  u.q[1] = *(const i4*)(base + 16);
  return u.v;
}

// Native-transcendental activations: v_exp_f32 + v_rcp_f32 (trans ops that
// co-execute with the XDL pipe), instead of precise-division expansions.
#define LOG2E 1.4426950408889634f
static __device__ inline float sigmoid_fast(float x) {
  return __builtin_amdgcn_rcpf(1.0f + __builtin_amdgcn_exp2f(-LOG2E * x));
}
static __device__ inline float tanh_fast(float x) {
  // tanh(x) = 2*sigmoid(2x) - 1
  return 2.0f * __builtin_amdgcn_rcpf(1.0f + __builtin_amdgcn_exp2f(-2.0f * LOG2E * x)) - 1.0f;
}

// ---------------------------------------------------------------------------
// Pack a row-major fp32 weight [Nrows x K] into bf16 WMMA B-fragments.
// Fragment fi = nt*kTiles + kt covers B[k = kt*32 .. +31][n = nt*16 .. +15],
// where B[k][n] = src[n][k] (we multiply by W^T).
// ---------------------------------------------------------------------------
__global__ void pack_bfrag_kernel(const float* __restrict__ src,
                                  __bf16* __restrict__ dst,
                                  int K, int kTiles, int total) {
  int id = blockIdx.x * blockDim.x + threadIdx.x;
  if (id >= total) return;
  int lane = id & 31;
  int p    = (id >> 5) & 1;
  int fi   = id >> 6;
  int kt   = fi % kTiles;
  int nt   = fi / kTiles;
  int n     = nt * 16 + (lane & 15);
  int kbase = kt * 32 + 8 * (lane >> 4) + 16 * p;
  __bf16* o = dst + (size_t)fi * 512 + p * 256 + lane * 8;
#pragma unroll
  for (int q = 0; q < 8; ++q)
    o[q] = (__bf16)src[(size_t)n * K + kbase + q];
}

// ---------------------------------------------------------------------------
// x_gates = C @ W_ih.T + b_ih + b_hh   -> fp32 [BATCH][G4] in workspace.
// ---------------------------------------------------------------------------
__global__ __launch_bounds__(256)
void xgates_kernel(const float* __restrict__ C,
                   const __bf16* __restrict__ wihF,
                   const float* __restrict__ b_ih,
                   const float* __restrict__ b_hh,
                   float* __restrict__ xg) {
  __shared__ __bf16 cbuf[16 * PITCHH];
  int tid = threadIdx.x, lane = tid & 31, w = tid >> 5, blk = blockIdx.x;
  int hf = lane >> 4, nl = lane & 15;

  for (int i = tid; i < 16 * HH; i += 256) {
    int m = i >> 8, k = i & 255;
    cbuf[m * PITCHH + k] = (__bf16)C[(size_t)(blk * 16 + m) * HH + k];
  }
  __syncthreads();

  v16bf af[8];
#pragma unroll
  for (int kt = 0; kt < 8; ++kt) af[kt] = load_afrag(cbuf, kt, lane);

  // 8 independent accumulators, kt-outer so consecutive WMMAs are independent.
  v8f acc[8];
#pragma unroll
  for (int idx = 0; idx < 8; ++idx) {
    int col = (idx * 8 + w) * 16 + nl;
    float bias = b_ih[col] + b_hh[col];
#pragma unroll
    for (int r = 0; r < 8; ++r) acc[idx][r] = bias;
  }
#pragma unroll
  for (int kt = 0; kt < 8; ++kt)
#pragma unroll
    for (int idx = 0; idx < 8; ++idx) {
      int nt = idx * 8 + w;
      v16bf b = load_bfrag(wihF, nt * 8 + kt, lane);
      acc[idx] = wmma_bf16(af[kt], b, acc[idx]);
    }
#pragma unroll
  for (int idx = 0; idx < 8; ++idx) {
    int col = (idx * 8 + w) * 16 + nl;
#pragma unroll
    for (int r = 0; r < 8; ++r)
      xg[(size_t)(blk * 16 + r + 8 * hf) * G4 + col] = acc[idx][r];
  }
}

// ---------------------------------------------------------------------------
// Main recurrent kernel. WG = 16 batch rows, 8 waves.
// Wave w owns gate columns [32w, 32w+32) of each i/f/g/o block:
//   N-tiles nt = g*16 + 2w + j (g=0..3 gate, j=0..1) -> pointwise is local.
// Output projection: wave w owns output columns [16w, 16w+16).
// ---------------------------------------------------------------------------
__global__ __launch_bounds__(256)
void lstm_main_kernel(const float* __restrict__ xgates,
                      const __bf16* __restrict__ whhF,
                      const __bf16* __restrict__ woutF,
                      const float* __restrict__ b_out,
                      float* __restrict__ out) {
  __shared__ __bf16 hbuf[16 * PITCHH];
  int tid = threadIdx.x, lane = tid & 31, w = tid >> 5, blk = blockIdx.x;
  int hf = lane >> 4, nl = lane & 15;
  int m0 = blk * 16;

  // h(0) = 0 in LDS
  for (int i = tid; i < 16 * PITCHH; i += 256) hbuf[i] = (__bf16)0.0f;

  // Step-invariant x_gates accumulator fragments (D-layout: VGPR r ->
  // m = r + 8*(lane>>4), n = lane&15).
  v8f xg[8];
#pragma unroll
  for (int g = 0; g < 4; ++g)
#pragma unroll
    for (int j = 0; j < 2; ++j) {
      int nt = g * 16 + 2 * w + j;
      int col = nt * 16 + nl;
      v8f a;
#pragma unroll
      for (int r = 0; r < 8; ++r)
        a[r] = xgates[(size_t)(m0 + r + 8 * hf) * G4 + col];
      xg[g * 2 + j] = a;
    }

  // W_out fragments for this wave's output tile: resident in registers.
  v16bf wof[8];
#pragma unroll
  for (int kt = 0; kt < 8; ++kt) wof[kt] = load_bfrag(woutF, w * 8 + kt, lane);
  float bo = b_out[w * 16 + nl];

  v8f c0 = {}, c1 = {};           // cell state, 32 cols x 16 rows per wave
  __syncthreads();

  // A-fragments of h(0) (zeros)
  v16bf af[8];
#pragma unroll
  for (int kt = 0; kt < 8; ++kt) af[kt] = load_afrag(hbuf, kt, lane);

  for (int t = 0; t < TSTEPS; ++t) {
    // ---- gates = x_gates + h @ W_hh.T  (64 wmma/wave) ----
    // kt-outer / tile-inner: consecutive WMMAs hit 8 independent
    // accumulators -> full XDL pipelining; af[kt] reused 8x per kt.
    v8f acc[8];
#pragma unroll
    for (int q = 0; q < 8; ++q) acc[q] = xg[q];
#pragma unroll
    for (int kt = 0; kt < 8; ++kt)
#pragma unroll
      for (int g = 0; g < 4; ++g)
#pragma unroll
        for (int j = 0; j < 2; ++j) {
          int nt = g * 16 + 2 * w + j;
          v16bf b = load_bfrag(whhF, nt * 8 + kt, lane);   // L2-resident stream
          acc[g * 2 + j] = wmma_bf16(af[kt], b, acc[g * 2 + j]);
        }

    // ---- LSTM cell (all tiles share the D-matrix lane layout) ----
    v8f hnew[2];
#pragma unroll
    for (int j = 0; j < 2; ++j) {
      v8f iv = acc[0 + j], fv = acc[2 + j], gv = acc[4 + j], ov = acc[6 + j];
      v8f& c = j ? c1 : c0;
      v8f hn;
#pragma unroll
      for (int r = 0; r < 8; ++r) {
        float ig = sigmoid_fast(iv[r]);
        float fg = sigmoid_fast(fv[r]);
        float og = sigmoid_fast(ov[r]);
        float gt = tanh_fast(gv[r]);
        float cn = fg * c[r] + ig * gt;
        c[r] = cn;
        hn[r] = og * tanh_fast(cn);
      }
      hnew[j] = hn;
    }

    // ---- publish h_new to LDS (bf16) ----
    __syncthreads();   // all waves done reading previous h fragments
#pragma unroll
    for (int j = 0; j < 2; ++j)
#pragma unroll
      for (int r = 0; r < 8; ++r)
        hbuf[(r + 8 * hf) * PITCHH + (32 * w + 16 * j + nl)] = (__bf16)hnew[j][r];
    __syncthreads();   // h_new visible to all waves

    // ---- reload A-fragments (h_new): used for y now and gates next step ----
#pragma unroll
    for (int kt = 0; kt < 8; ++kt) af[kt] = load_afrag(hbuf, kt, lane);

    // ---- y = h_new @ W_out.T + b_out  (8 wmma/wave) ----
    v8f y;
#pragma unroll
    for (int r = 0; r < 8; ++r) y[r] = bo;
#pragma unroll
    for (int kt = 0; kt < 8; ++kt) y = wmma_bf16(af[kt], wof[kt], y);

    size_t obase = ((size_t)t * BATCH + m0) * OUT;
#pragma unroll
    for (int r = 0; r < 8; ++r)
      out[obase + (size_t)(r + 8 * hf) * OUT + w * 16 + nl] = y[r];
  }
}

// ---------------------------------------------------------------------------
// Launch. Inputs: C, W_ih, W_hh, b_ih, b_hh, W_out, b_out, max_seq_len(=100).
// Workspace layout: whhF(512K) | wihF(512K) | woutF(64K) | xgates(8M).
// ---------------------------------------------------------------------------
extern "C" void kernel_launch(void* const* d_in, const int* in_sizes, int n_in,
                              void* d_out, int out_size, void* d_ws, size_t ws_size,
                              hipStream_t stream) {
  const float* C     = (const float*)d_in[0];
  const float* W_ih  = (const float*)d_in[1];
  const float* W_hh  = (const float*)d_in[2];
  const float* b_ih  = (const float*)d_in[3];
  const float* b_hh  = (const float*)d_in[4];
  const float* W_out = (const float*)d_in[5];
  const float* b_out = (const float*)d_in[6];
  float* out = (float*)d_out;

  char* ws = (char*)d_ws;
  __bf16* whhF  = (__bf16*)(ws);                       // 64*8 frags * 1KB
  __bf16* wihF  = (__bf16*)(ws + (512 << 10));
  __bf16* woutF = (__bf16*)(ws + (1024 << 10));        // 8*8 frags * 1KB
  float*  xg    = (float*)(ws + (1024 << 10) + (64 << 10));

  // Pack weights -> bf16 B-fragments.
  {
    int total = 64 * 8 * 64;  // W_hh / W_ih: nTiles=64, kTiles=8, 64 chunks/frag
    int blocks = (total + 255) / 256;
    pack_bfrag_kernel<<<blocks, 256, 0, stream>>>(W_hh, whhF, HH, 8, total);
    pack_bfrag_kernel<<<blocks, 256, 0, stream>>>(W_ih, wihF, HH, 8, total);
    int totalO = 8 * 8 * 64;  // W_out: nTiles=8
    pack_bfrag_kernel<<<(totalO + 255) / 256, 256, 0, stream>>>(W_out, woutF, HH, 8, totalO);
  }

  // x_gates (once).
  xgates_kernel<<<BATCH / 16, 256, 0, stream>>>(C, wihF, b_ih, b_hh, xg);

  // Recurrence (T=100 inside the kernel, batch-parallel across 128 WGs).
  lstm_main_kernel<<<BATCH / 16, 256, 0, stream>>>(xg, whhF, woutF, b_out, out);
}
